// ModernBERTAttention_51883204935812
// MI455X (gfx1250) — matrix operand order repaired and checked by reading, not compiled, not run on hardware
//
#include <hip/hip_runtime.h>

// ---------------------------------------------------------------------------
// ModernBERT attention on gfx1250 (MI455X): bf16 WMMA (v_wmma_f32_16x16x32_bf16)
// B=4, S=2048, E=1024, H=16, D=64.  Compute-bound (~137 GFLOP vs <0.5 GB HBM).
// Attention stages K/V tiles in LDS via GLOBAL_LOAD_ASYNC_TO_LDS_B128
// (ASYNCcnt) shared by all 8 waves of a workgroup.
// ---------------------------------------------------------------------------

typedef __bf16 bf16;
typedef __attribute__((ext_vector_type(16))) __bf16 v16bf;
typedef __attribute__((ext_vector_type(8)))  float  v8f;
typedef __attribute__((ext_vector_type(4)))  unsigned int u32x4;

#define B_ 4
#define S_ 2048
#define E_ 1024
#define H_ 16
#define D_ 64
#define QKVN_ 3072
#define M_ (B_ * S_)   // 8192 rows

// -------------------------- WMMA helpers -----------------------------------

__device__ __forceinline__ v8f wmma_bf16(v16bf a, v16bf b, v8f c) {
  // 8 args: (neg_a, A, neg_b, B, c_mod, C, reuse_a, reuse_b)
  return __builtin_amdgcn_wmma_f32_16x16x32_bf16(false, a, false, b, (short)0, c,
                                                 false, false);
}

// A-matrix 16x32 bf16 fragment from row-major [16+][ld] source.
// ISA layout: lanes 0-15 row=lane, K 0..7 (V0..3) and 16..23 (V4..7);
//             lanes 16-31 row=lane-16, K 8..15 and 24..31.
// => per lane: two contiguous 16-byte runs at k = hf*8 and k = 16+hf*8.
__device__ __forceinline__ v16bf load_frag_a(const bf16* base, int ld) {
  int lane = threadIdx.x & 31;
  int row  = lane & 15;
  int hf   = lane >> 4;
  const bf16* p = base + (size_t)row * ld + hf * 8;
  union { v16bf v; u32x4 q[2]; } f;
  f.q[0] = *(const u32x4*)(p);
  f.q[1] = *(const u32x4*)(p + 16);
  return f.v;
}

// B-matrix 32x16 bf16 fragment from K-contiguous [N][ld] source (pre-transposed).
// ISA layout: lanes 0-15 N=lane hold K 0..7,16..23; lanes 16-31 N=lane-16 hold K 8..15,24..31.
// => per lane: two contiguous 16-byte runs at k = hf*8 and k = 16+hf*8.
__device__ __forceinline__ v16bf load_frag_b(const bf16* baseT, int ld) {
  int lane = threadIdx.x & 31;
  int n    = lane & 15;
  int hf   = lane >> 4;
  const bf16* p = baseT + (size_t)n * ld + hf * 8;
  union { v16bf v; u32x4 q[2]; } f;
  f.q[0] = *(const u32x4*)(p);
  f.q[1] = *(const u32x4*)(p + 16);
  return f.v;
}

__device__ __forceinline__ void lds_fence() {
  // CDNA5 split dependency counter: make intra-wave LDS stores visible to the
  // following ds_load_b128 fragment loads (cross-lane data exchange).
  asm volatile("s_wait_dscnt 0" ::: "memory");
}

// CDNA5 async memory->LDS copy (ASYNCcnt-tracked; GV addressing mode).
// ldsOff: workgroup-relative LDS byte address (VDST); gaddr: 64-bit global.
__device__ __forceinline__ void async_copy_b128(unsigned ldsOff, const void* gaddr) {
  asm volatile("global_load_async_to_lds_b128 %0, %1, off"
               :: "v"(ldsOff), "v"(gaddr) : "memory");
}

__device__ __forceinline__ void wait_asynccnt0() {
  asm volatile("s_wait_asynccnt 0x0" ::: "memory");
}

// low 32 bits of the generic address = workgroup-relative LDS byte address
__device__ __forceinline__ unsigned lds_addr32(const void* p) {
  return (unsigned)(size_t)p;
}

// -------------------------- prep kernels -----------------------------------

__global__ void cvt_f32_bf16_kernel(const float* __restrict__ in,
                                    bf16* __restrict__ out, int n) {
  int i = blockIdx.x * blockDim.x + threadIdx.x;
  if (i < n) out[i] = (bf16)in[i];
}

// W row-major [K][N] f32  ->  WT [N][K] bf16 (K contiguous)
__global__ void transpose_cvt_kernel(const float* __restrict__ W,
                                     bf16* __restrict__ WT, int K, int N) {
  int i = blockIdx.x * blockDim.x + threadIdx.x;
  if (i >= K * N) return;
  int k = i % K;
  int n = i / K;
  WT[(size_t)n * K + k] = (bf16)W[(size_t)k * N + n];
}

// -------------------------- bf16 WMMA GEMM ---------------------------------
// C[M][N] f32 = A[M][K] bf16 (row-major) x BT[N][K] bf16 (K contiguous).
// One wave computes a 32x64 tile: 2 A-frags x 4 B-frags -> 8 WMMAs / K-step.
__global__ void gemm_bf16_kernel(const bf16* __restrict__ A,
                                 const bf16* __restrict__ BT,
                                 float* __restrict__ C, int M, int N, int K) {
  int wave = (int)((blockIdx.x * blockDim.x + threadIdx.x) >> 5);
  int numM = M >> 5, numN = N >> 6;
  if (wave >= numM * numN) return;
  int mt = wave % numM;
  int nt = wave / numM;
  int lane = threadIdx.x & 31;
  int hf = lane >> 4, lr = lane & 15;

  v8f acc[2][4];
#pragma unroll
  for (int i = 0; i < 2; ++i)
#pragma unroll
    for (int j = 0; j < 4; ++j)
#pragma unroll
      for (int v = 0; v < 8; ++v) acc[i][j][v] = 0.0f;

  const bf16* Abase = A + (size_t)(mt * 32) * K;
  const bf16* Bbase = BT + (size_t)(nt * 64) * K;

  for (int k0 = 0; k0 < K; k0 += 32) {
    v16bf a[2], b[4];
    a[0] = load_frag_a(Abase + k0, K);
    a[1] = load_frag_a(Abase + (size_t)16 * K + k0, K);
#pragma unroll
    for (int j = 0; j < 4; ++j)
      b[j] = load_frag_b(Bbase + (size_t)(j * 16) * K + k0, K);
#pragma unroll
    for (int i = 0; i < 2; ++i)
#pragma unroll
      for (int j = 0; j < 4; ++j) acc[i][j] = wmma_bf16(a[i], b[j], acc[i][j]);
  }

#pragma unroll
  for (int i = 0; i < 2; ++i)
#pragma unroll
    for (int j = 0; j < 4; ++j)
#pragma unroll
      for (int v = 0; v < 8; ++v) {
        int row = mt * 32 + i * 16 + hf * 8 + v;
        int col = nt * 64 + j * 16 + lr;
        C[(size_t)row * N + col] = acc[i][j][v];
      }
}

// -------------------------- QKV split + RoPE -------------------------------
// qkv f32 [B*S][3072] -> Qb,Kb bf16 [b][h][s][64] (RoPE applied),
//                        Vt bf16 [b][h][64][s]   (transposed for PV B-frags).
__global__ void qkv_post_kernel(const float* __restrict__ qkv,
                                bf16* __restrict__ Qb, bf16* __restrict__ Kb,
                                bf16* __restrict__ Vt) {
  int gid = blockIdx.x * blockDim.x + threadIdx.x;   // B*H*S*32 = 2^22 threads
  if (gid >= B_ * H_ * S_ * 32) return;
  int i = gid & 31;            // rotary pair index (d = i, i+32)
  int s = (gid >> 5) & (S_ - 1);
  int h = (gid >> 16) & (H_ - 1);
  int b = gid >> 20;

  const float* row = qkv + (size_t)(b * S_ + s) * QKVN_ + h * D_;
  float q1 = row[i],        q2 = row[i + 32];
  float k1 = row[E_ + i],   k2 = row[E_ + i + 32];
  float v1 = row[2*E_ + i], v2 = row[2*E_ + i + 32];

  float freq = __powf(10000.0f, -(float)(2 * i) * (1.0f / 64.0f));
  float ang  = (float)s * freq;
  float sn, cs;
  __sincosf(ang, &sn, &cs);

  size_t qko = ((size_t)(b * H_ + h) * S_ + s) * D_;
  Qb[qko + i]      = (bf16)(q1 * cs - q2 * sn);
  Qb[qko + i + 32] = (bf16)(q2 * cs + q1 * sn);
  Kb[qko + i]      = (bf16)(k1 * cs - k2 * sn);
  Kb[qko + i + 32] = (bf16)(k2 * cs + k1 * sn);

  size_t vto = (size_t)(b * H_ + h) * D_ * S_ + s;
  Vt[vto + (size_t)i * S_]        = (bf16)v1;
  Vt[vto + (size_t)(i + 32) * S_] = (bf16)v2;
}

// -------------------------- flash attention --------------------------------
// One block (8 waves) per (b, h, group of 8 query tiles): the 8 waves share
// each 32-key K/V tile, cooperatively staged into LDS with
// global_load_async_to_lds_b128 (ASYNCcnt) + s_wait_asynccnt + split barrier.
// Per wave: online softmax in C-layout (row = vgpr idx, col = lane) via
// 16-lane butterfly shuffles; P re-layout C-tile -> A-frag through LDS.
__global__ void attn_kernel(const bf16* __restrict__ Qb,
                            const bf16* __restrict__ Kb,
                            const bf16* __restrict__ Vt,
                            bf16* __restrict__ Obf) {
  __shared__ __align__(16) bf16 Ktile[32 * 64];    // [key][d]     4 KB
  __shared__ __align__(16) bf16 Vtile[64 * 32];    // [d][key]     4 KB
  __shared__ __align__(16) bf16 Plds[8][16 * 32];  // per-wave P   8 KB

  int t    = threadIdx.x;          // 0..255
  int wib  = t >> 5;               // wave in block: 0..7
  int blk  = blockIdx.x;           // B*H*16 blocks, exact
  int qt   = (blk & 15) * 8 + wib; // query tile 0..127
  int h    = (blk >> 4) & (H_ - 1);
  int b    = blk >> 8;
  int lane = t & 31;
  int hf = lane >> 4, lr = lane & 15;
  const float scale = 0.125f;      // 1/sqrt(64)

  const bf16* Qt = Qb + ((size_t)(b * H_ + h) * S_ + qt * 16) * D_;
  const bf16* Kh = Kb + (size_t)(b * H_ + h) * S_ * D_;
  const bf16* Vh = Vt + (size_t)(b * H_ + h) * D_ * S_;

  v16bf aq0 = load_frag_a(Qt, D_);        // d 0..31
  v16bf aq1 = load_frag_a(Qt + 32, D_);   // d 32..63

  v8f o[4];
#pragma unroll
  for (int j = 0; j < 4; ++j)
#pragma unroll
    for (int v = 0; v < 8; ++v) o[j][v] = 0.0f;
  float m[8], l[8];
#pragma unroll
  for (int v = 0; v < 8; ++v) { m[v] = -1e30f; l[v] = 0.0f; }

  bf16* P = &Plds[wib][0];
  v8f z;
#pragma unroll
  for (int v = 0; v < 8; ++v) z[v] = 0.0f;

  unsigned kLds = lds_addr32(&Ktile[0]) + (unsigned)t * 16u;
  unsigned vLds = lds_addr32(&Vtile[0]) + (unsigned)t * 16u;
  int vd = t >> 2, vc = t & 3;     // V tile: row d = t/4, 16B chunk t%4

  for (int k2 = 0; k2 < S_; k2 += 32) {
    // cooperative async staging: 32 keys x 64 d (K) and 64 d x 32 keys (V)
    async_copy_b128(kLds, (const char*)(Kh + (size_t)k2 * D_) + (size_t)t * 16);
    async_copy_b128(vLds, (const char*)(Vh + (size_t)vd * S_ + k2) + vc * 16);
    wait_asynccnt0();
    __syncthreads();

    // scores for 32 keys: two 16x16 tiles, each 2 WMMAs over d (K=64)
    v16bf b00 = load_frag_b(Ktile, D_);
    v16bf b01 = load_frag_b(Ktile + 32, D_);
    v16bf b10 = load_frag_b(Ktile + 16 * D_, D_);
    v16bf b11 = load_frag_b(Ktile + 16 * D_ + 32, D_);
    v8f s0 = wmma_bf16(aq0, b00, z);
    s0     = wmma_bf16(aq1, b01, s0);
    v8f s1 = wmma_bf16(aq0, b10, z);
    s1     = wmma_bf16(aq1, b11, s1);

#pragma unroll
    for (int v = 0; v < 8; ++v) {
      float x0 = s0[v] * scale, x1 = s1[v] * scale;
      float sm = fmaxf(x0, x1);
      sm = fmaxf(sm, __shfl_xor(sm, 8, 16));
      sm = fmaxf(sm, __shfl_xor(sm, 4, 16));
      sm = fmaxf(sm, __shfl_xor(sm, 2, 16));
      sm = fmaxf(sm, __shfl_xor(sm, 1, 16));
      float mn = fmaxf(m[v], sm);
      float p0 = __expf(x0 - mn), p1 = __expf(x1 - mn);
      float rs = p0 + p1;
      rs += __shfl_xor(rs, 8, 16);
      rs += __shfl_xor(rs, 4, 16);
      rs += __shfl_xor(rs, 2, 16);
      rs += __shfl_xor(rs, 1, 16);
      float corr = __expf(m[v] - mn);
      l[v] = l[v] * corr + rs;
      m[v] = mn;
#pragma unroll
      for (int j = 0; j < 4; ++j) o[j][v] *= corr;
      int prow = hf * 8 + v;
      P[prow * 32 + lr]      = (bf16)p0;
      P[prow * 32 + 16 + lr] = (bf16)p1;
    }

    lds_fence();  // intra-wave cross-lane exchange of P
    v16bf pf = load_frag_a(P, 32);  // P as 16x32 A-matrix (keys = K dim)
#pragma unroll
    for (int j = 0; j < 4; ++j) {
      v16bf bv = load_frag_b(Vtile + (j * 16) * 32, 32);
      o[j] = wmma_bf16(pf, bv, o[j]);
    }

    __syncthreads();  // all waves done with K/V tiles before next staging
  }

  // normalize and store O as bf16 [b][s][h*64+d] for the output GEMM
#pragma unroll
  for (int v = 0; v < 8; ++v) {
    float inv = 1.0f / l[v];
    int srow = qt * 16 + hf * 8 + v;
    size_t orow = ((size_t)b * S_ + srow) * E_ + h * D_;
#pragma unroll
    for (int j = 0; j < 4; ++j)
      Obf[orow + j * 16 + lr] = (bf16)(o[j][v] * inv);
  }
}

// -------------------------- launch -----------------------------------------

extern "C" void kernel_launch(void* const* d_in, const int* in_sizes, int n_in,
                              void* d_out, int out_size, void* d_ws, size_t ws_size,
                              hipStream_t stream) {
  const float* x    = (const float*)d_in[0];  // [4][2048][1024]
  const float* wqkv = (const float*)d_in[1];  // [1024][3072]
  const float* wo   = (const float*)d_in[2];  // [1024][1024]
  float* out = (float*)d_out;                 // [4][2048][1024]

  char* ws = (char*)d_ws;
  // workspace layout (bytes)
  bf16*  xb     = (bf16*)(ws + 0);                    //  16 MB  x bf16
  bf16*  wqkvT  = (bf16*)(ws + 16777216);             //   6 MB  wqkv^T bf16
  bf16*  woT    = (bf16*)(ws + 23068672);             //   2 MB  wo^T bf16
  float* qkvf   = (float*)(ws + 25165824);            //  96 MB  qkv f32
  bf16*  Qb     = (bf16*)(ws + 125829120);            //  16 MB
  bf16*  Kb     = (bf16*)(ws + 142606336);            //  16 MB
  bf16*  Vt     = (bf16*)(ws + 159383552);            //  16 MB
  bf16*  Obf    = (bf16*)(ws + 176160768);            //  16 MB  (end ~184 MB)

  const int TPB = 256;

  // 1) precision conversion / weight transposition
  {
    int n = M_ * E_;  // 8388608
    cvt_f32_bf16_kernel<<<(n + TPB - 1) / TPB, TPB, 0, stream>>>(x, xb, n);
  }
  {
    int n = E_ * QKVN_;
    transpose_cvt_kernel<<<(n + TPB - 1) / TPB, TPB, 0, stream>>>(wqkv, wqkvT, E_, QKVN_);
  }
  {
    int n = E_ * E_;
    transpose_cvt_kernel<<<(n + TPB - 1) / TPB, TPB, 0, stream>>>(wo, woT, E_, E_);
  }

  // 2) QKV projection: [8192x1024] x [1024x3072]
  {
    int waves = (M_ / 32) * (QKVN_ / 64);  // 12288
    int blocks = (waves * 32 + TPB - 1) / TPB;
    gemm_bf16_kernel<<<blocks, TPB, 0, stream>>>(xb, wqkvT, qkvf, M_, QKVN_, E_);
  }

  // 3) split + RoPE + V transpose
  {
    int n = B_ * H_ * S_ * 32;
    qkv_post_kernel<<<(n + TPB - 1) / TPB, TPB, 0, stream>>>(qkvf, Qb, Kb, Vt);
  }

  // 4) flash attention: one block (8 waves) per (b, h, 8 query tiles)
  {
    int blocks = B_ * H_ * 16;  // 1024, exact
    attn_kernel<<<blocks, TPB, 0, stream>>>(Qb, Kb, Vt, Obf);
  }

  // 5) output projection: [8192x1024] x [1024x1024] -> f32 out
  {
    int waves = (M_ / 32) * (E_ / 64);  // 4096
    int blocks = (waves * 32 + TPB - 1) / TPB;
    gemm_bf16_kernel<<<blocks, TPB, 0, stream>>>(Obf, woT, out, M_, E_, E_);
  }

  (void)in_sizes; (void)n_in; (void)out_size; (void)ws_size;
}
